// BitNetLinear_80642305950374
// MI455X (gfx1250) — 8-bit matrix operand order repaired and checked by reading, compiled, not run on hardware
//
#include <hip/hip_runtime.h>
#include <hip/hip_bf16.h>
#include <math.h>

typedef __attribute__((ext_vector_type(8))) int v8i;

#define MDIM 16384
#define NDIM 2048
#define KDIM 2048

__global__ void __launch_bounds__(256) quant_x_kernel(const float* __restrict__ x,
                                                      signed char* __restrict__ xq,
                                                      float* __restrict__ xscale) {
    __shared__ float red[256];
    const int row = blockIdx.x;
    const int t   = threadIdx.x;
    const float* rp = x + (size_t)row * KDIM + (size_t)t * 8;
    float4 v0 = ((const float4*)rp)[0];
    float4 v1 = ((const float4*)rp)[1];
    float m = fmaxf(fmaxf(fmaxf(fabsf(v0.x), fabsf(v0.y)), fmaxf(fabsf(v0.z), fabsf(v0.w))),
                    fmaxf(fmaxf(fabsf(v1.x), fabsf(v1.y)), fmaxf(fabsf(v1.z), fabsf(v1.w))));
    red[t] = m;
    __syncthreads();
    for (int s = 128; s > 0; s >>= 1) {
        if (t < s) red[t] = fmaxf(red[t], red[t + s]);
        __syncthreads();
    }
    const float scale = fmaxf(red[0], 1e-5f);
    if (t == 0) xscale[row] = scale;
    const float inv = 127.0f / scale;
    const float vals[8] = {v0.x, v0.y, v0.z, v0.w, v1.x, v1.y, v1.z, v1.w};
    int lo = 0, hi = 0;
#pragma unroll
    for (int j = 0; j < 4; ++j) {
        int q = (int)rintf(vals[j] * inv);
        q = max(-128, min(127, q));
        lo |= (q & 0xff) << (8 * j);
    }
#pragma unroll
    for (int j = 0; j < 4; ++j) {
        int q = (int)rintf(vals[4 + j] * inv);
        q = max(-128, min(127, q));
        hi |= (q & 0xff) << (8 * j);
    }
    *(int2*)(xq + (size_t)row * KDIM + (size_t)t * 8) = make_int2(lo, hi);
}

__global__ void __launch_bounds__(256) wsum_kernel(const float* __restrict__ w,
                                                   float* __restrict__ partial) {
    __shared__ float red[256];
    const int t = threadIdx.x;
    const size_t total = (size_t)NDIM * KDIM;
    float s = 0.f;
    for (size_t i = (size_t)blockIdx.x * 256 + t; i < total; i += 256 * 256) s += w[i];
    red[t] = s;
    __syncthreads();
    for (int st = 128; st > 0; st >>= 1) {
        if (t < st) red[t] += red[t + st];
        __syncthreads();
    }
    if (t == 0) partial[blockIdx.x] = red[0];
}

__global__ void __launch_bounds__(256) wmean_kernel(const float* __restrict__ partial,
                                                    float* __restrict__ stats) {
    __shared__ float red[256];
    const int t = threadIdx.x;
    red[t] = partial[t];
    __syncthreads();
    for (int st = 128; st > 0; st >>= 1) {
        if (t < st) red[t] += red[t + st];
        __syncthreads();
    }
    if (t == 0) stats[0] = red[0] / (float)((size_t)NDIM * KDIM);
}

__global__ void __launch_bounds__(256) wabs_kernel(const float* __restrict__ w,
                                                   const float* __restrict__ stats,
                                                   float* __restrict__ partial) {
    __shared__ float red[256];
    const int t = threadIdx.x;
    const float mean = stats[0];
    const size_t total = (size_t)NDIM * KDIM;
    float s = 0.f;
    for (size_t i = (size_t)blockIdx.x * 256 + t; i < total; i += 256 * 256)
        s += fabsf(w[i] - mean);
    red[t] = s;
    __syncthreads();
    for (int st = 128; st > 0; st >>= 1) {
        if (t < st) red[t] += red[t + st];
        __syncthreads();
    }
    if (t == 0) partial[blockIdx.x] = red[0];
}

__global__ void __launch_bounds__(256) wscale_kernel(const float* __restrict__ partial,
                                                     float* __restrict__ stats) {
    __shared__ float red[256];
    const int t = threadIdx.x;
    red[t] = partial[t];
    __syncthreads();
    for (int st = 128; st > 0; st >>= 1) {
        if (t < st) red[t] += red[t + st];
        __syncthreads();
    }
    if (t == 0) stats[1] = fmaxf(red[0] / (float)((size_t)NDIM * KDIM), 1e-5f);
}

__global__ void __launch_bounds__(256) quant_w_kernel(const float* __restrict__ w,
                                                      const float* __restrict__ stats,
                                                      signed char* __restrict__ wq) {
    const float mean = stats[0];
    const float inv  = 1.0f / stats[1];
    const size_t base = ((size_t)blockIdx.x * 256 + threadIdx.x) * 8;
    float4 v0 = ((const float4*)(w + base))[0];
    float4 v1 = ((const float4*)(w + base))[1];
    const float vals[8] = {v0.x, v0.y, v0.z, v0.w, v1.x, v1.y, v1.z, v1.w};
    int lo = 0, hi = 0;
#pragma unroll
    for (int j = 0; j < 4; ++j) {
        int q = (int)rintf((vals[j] - mean) * inv);
        q = max(-1, min(1, q));
        lo |= (q & 0xff) << (8 * j);
    }
#pragma unroll
    for (int j = 0; j < 4; ++j) {
        int q = (int)rintf((vals[4 + j] - mean) * inv);
        q = max(-1, min(1, q));
        hi |= (q & 0xff) << (8 * j);
    }
    *(int2*)(wq + base) = make_int2(lo, hi);
}

__global__ void __launch_bounds__(256) gemm_iu8_kernel(const signed char* __restrict__ xq,
                                                       const signed char* __restrict__ wq,
                                                       const float* __restrict__ xscale,
                                                       const float* __restrict__ stats,
                                                       float* __restrict__ y) {
    const int lane = threadIdx.x & 31;
    const int wv   = threadIdx.x >> 5;
    const int half = lane >> 4;
    const int l16  = lane & 15;

    const int m_base = blockIdx.y * 128 + (wv >> 2) * 64;
    const int n_base = blockIdx.x * 128 + (wv & 3) * 32;

    const signed char* arow[4];
#pragma unroll
    for (int t = 0; t < 4; ++t)
        arow[t] = xq + (size_t)(m_base + t * 16 + l16) * KDIM;
    const signed char* brow[2];
#pragma unroll
    for (int u = 0; u < 2; ++u)
        brow[u] = wq + (size_t)(n_base + u * 16 + l16) * KDIM;

    v8i acc[4][2];
#pragma unroll
    for (int t = 0; t < 4; ++t)
#pragma unroll
        for (int u = 0; u < 2; ++u)
            acc[t][u] = v8i{0, 0, 0, 0, 0, 0, 0, 0};

    for (int k0 = 0; k0 < KDIM; k0 += 64) {
        v8i a[4], b[2];
#pragma unroll
        for (int t = 0; t < 4; ++t) {
            int2 q0 = *(const int2*)(arow[t] + k0 +      8 * half);
            int2 q1 = *(const int2*)(arow[t] + k0 + 16 + 8 * half);
            int2 q2 = *(const int2*)(arow[t] + k0 + 32 + 8 * half);
            int2 q3 = *(const int2*)(arow[t] + k0 + 48 + 8 * half);
            a[t][0] = q0.x; a[t][1] = q0.y;
            a[t][2] = q1.x; a[t][3] = q1.y;
            a[t][4] = q2.x; a[t][5] = q2.y;
            a[t][6] = q3.x; a[t][7] = q3.y;
        }
#pragma unroll
        for (int u = 0; u < 2; ++u) {
            int2 q0 = *(const int2*)(brow[u] + k0 +      8 * half);
            int2 q1 = *(const int2*)(brow[u] + k0 + 16 + 8 * half);
            int2 q2 = *(const int2*)(brow[u] + k0 + 32 + 8 * half);
            int2 q3 = *(const int2*)(brow[u] + k0 + 48 + 8 * half);
            b[u][0] = q0.x; b[u][1] = q0.y; b[u][2] = q1.x; b[u][3] = q1.y;
            b[u][4] = q2.x; b[u][5] = q2.y; b[u][6] = q3.x; b[u][7] = q3.y;
        }
#pragma unroll
        for (int t = 0; t < 4; ++t)
#pragma unroll
            for (int u = 0; u < 2; ++u)
                acc[t][u] = __builtin_amdgcn_wmma_i32_16x16x64_iu8(
                     true, a[t],  true, b[u], acc[t][u],
                     false,  false);
    }

    const float ws127 = stats[1] * (1.0f / 127.0f);
#pragma unroll
    for (int t = 0; t < 4; ++t) {
#pragma unroll
        for (int r = 0; r < 8; ++r) {
            const int grow = m_base + t * 16 + r + 8 * half;
            const float s  = ws127 * xscale[grow];
#pragma unroll
            for (int u = 0; u < 2; ++u) {
                const int gcol = n_base + u * 16 + l16;
                y[(size_t)grow * NDIM + gcol] = s * (float)acc[t][u][r];
            }
        }
    }
}

extern "C" void kernel_launch(void* const* d_in, const int* in_sizes, int n_in,
                              void* d_out, int out_size, void* d_ws, size_t ws_size,
                              hipStream_t stream) {
    const float* x = (const float*)d_in[0];
    const float* w = (const float*)d_in[1];
    float* y = (float*)d_out;

    char* ws = (char*)d_ws;
    const size_t XQ_BYTES = (size_t)MDIM * KDIM;
    const size_t XS_BYTES = (size_t)MDIM * sizeof(float);
    const size_t WQ_BYTES = (size_t)NDIM * KDIM;
    signed char* xq   = (signed char*)(ws);
    float* xscale     = (float*)(ws + XQ_BYTES);
    signed char* wq   = (signed char*)(ws + XQ_BYTES + XS_BYTES);
    float* partial    = (float*)(ws + XQ_BYTES + XS_BYTES + WQ_BYTES);
    float* stats      = (float*)(ws + XQ_BYTES + XS_BYTES + WQ_BYTES + 256 * sizeof(float));

    quant_x_kernel<<<MDIM, 256, 0, stream>>>(x, xq, xscale);
    wsum_kernel<<<256, 256, 0, stream>>>(w, partial);
    wmean_kernel<<<1, 256, 0, stream>>>(partial, stats);
    wabs_kernel<<<256, 256, 0, stream>>>(w, stats, partial);
    wscale_kernel<<<1, 256, 0, stream>>>(partial, stats);
    quant_w_kernel<<<((size_t)NDIM * KDIM) / (256 * 8), 256, 0, stream>>>(w, stats, wq);
    dim3 grid(NDIM / 128, MDIM / 128);
    gemm_iu8_kernel<<<grid, 256, 0, stream>>>(xq, wq, xscale, stats, y);
}
